// SupCR_50921132261428
// MI455X (gfx1250) — matrix operand order repaired and checked by reading, not compiled, not run on hardware
//
#include <hip/hip_runtime.h>

typedef __attribute__((ext_vector_type(16))) __bf16 v16bf;
typedef __attribute__((ext_vector_type(8)))  __bf16 v8bf;
typedef __attribute__((ext_vector_type(8)))  float  v8f;

#define TWO_N   512
#define DMODEL  256
#define TEMP_F  1.0f

// workspace layout (bytes); total ~1.51 MB
#define OFF_EHI 0
#define OFF_ELO (TWO_N * DMODEL * 2)                 // 262144
#define OFF_NN  (2 * TWO_N * DMODEL * 2)             // 524288
#define OFF_S   (OFF_NN + TWO_N * 4)                 // 526336
#define OFF_ROW (OFF_S + TWO_N * TWO_N * 4)          // 1574912

// --- Kernel 0: split fp32 embeddings into bf16 hi/lo pair (3-product emulation)
__global__ void k_prep(const float* __restrict__ E,
                       __bf16* __restrict__ Ehi, __bf16* __restrict__ Elo) {
    int idx = blockIdx.x * blockDim.x + threadIdx.x;
    if (idx >= TWO_N * DMODEL) return;
    float v  = E[idx];
    __bf16 h = (__bf16)v;
    Ehi[idx] = h;
    Elo[idx] = (__bf16)(v - (float)h);
}

// --- Kernel 1: exact fp32 row norms n_i = ||e_i||^2
__global__ void k_norm(const float* __restrict__ E, float* __restrict__ nn) {
    int row = blockIdx.x * blockDim.x + threadIdx.x;
    if (row >= TWO_N) return;
    const float4* p = (const float4*)(E + (size_t)row * DMODEL);
    float acc = 0.0f;
    #pragma unroll 4
    for (int c = 0; c < DMODEL / 4; ++c) {
        float4 v = p[c];
        acc += v.x * v.x + v.y * v.y + v.z * v.z + v.w * v.w;
    }
    nn[row] = acc;
}

// --- Kernel 2: WMMA Gram GEMM fused with s = exp(-dist). One wave per 16x16 tile,
// 8 waves per block, 1024 tiles total. G = Hi*Hi^T + Hi*Lo^T + Lo*Hi^T (bf16 split).
__global__ void __launch_bounds__(256)
k_gemm_s(const __bf16* __restrict__ Ehi, const __bf16* __restrict__ Elo,
         const float* __restrict__ nn, float* __restrict__ S) {
    int wave = threadIdx.x >> 5;
    int lane = threadIdx.x & 31;
    int tile = blockIdx.x * 8 + wave;
    int i0 = (tile >> 5) * 16;        // 32 tiles per dimension
    int k0 = (tile & 31) * 16;
    int hf = lane >> 4;               // wave32 half
    int lr = lane & 15;

    const __bf16* aHi = Ehi + (size_t)(i0 + lr) * DMODEL;
    const __bf16* aLo = Elo + (size_t)(i0 + lr) * DMODEL;
    const __bf16* bHi = Ehi + (size_t)(k0 + lr) * DMODEL;
    const __bf16* bLo = Elo + (size_t)(k0 + lr) * DMODEL;

    v8f acc = {};
    for (int kk = 0; kk < DMODEL; kk += 32) {
        __builtin_prefetch(aHi + kk + 32, 0, 0);
        __builtin_prefetch(bHi + kk + 32, 0, 0);
        // A fragment (16x32 bf16): lanes 0-15 K=0..15, lanes 16-31 K=16..31
        v8bf c0 = *(const v8bf*)(aHi + kk + hf * 16);
        v8bf c1 = *(const v8bf*)(aHi + kk + 8 + hf * 16);
        v8bf l0 = *(const v8bf*)(aLo + kk + hf * 16);
        v8bf l1 = *(const v8bf*)(aLo + kk + 8 + hf * 16);
        v16bf ahi, alo;
        #pragma unroll
        for (int e = 0; e < 8; ++e) {
            ahi[e] = c0[e]; ahi[8 + e] = c1[e];
            alo[e] = l0[e]; alo[8 + e] = l1[e];
        }
        // B fragment (32x16 bf16): N=lane%16; lanes 0-15 K=kk+0..15, lanes 16-31 K=kk+16..31
        v16bf bhi = *(const v16bf*)(bHi + kk + hf * 16);
        v16bf blo = *(const v16bf*)(bLo + kk + hf * 16);

        acc = __builtin_amdgcn_wmma_f32_16x16x32_bf16(false, ahi, false, bhi,
                                                      (short)0, acc, false, false);
        acc = __builtin_amdgcn_wmma_f32_16x16x32_bf16(false, ahi, false, blo,
                                                      (short)0, acc, false, false);
        acc = __builtin_amdgcn_wmma_f32_16x16x32_bf16(false, alo, false, bhi,
                                                      (short)0, acc, false, false);
    }

    int k = k0 + lr;
    float nk = nn[k];
    #pragma unroll
    for (int r = 0; r < 8; ++r) {
        int i = i0 + hf * 8 + r;                        // C/D layout: M = hf*8 + r
        float g    = acc[r];
        float n2   = fmaxf(nn[i] + nk - 2.0f * g, 0.0f);
        float dist = sqrtf(n2);
        S[(size_t)i * TWO_N + k] = expf(-dist / TEMP_F);
    }
}

// --- Kernel 3: per-row masked denominators + log terms. One block per row i;
// s-row and d-row live in LDS, inner j-loop is a uniform LDS broadcast.
__global__ void __launch_bounds__(256)
k_denom(const float* __restrict__ S, const float* __restrict__ T,
        float* __restrict__ rowpart) {
    __shared__ float sh_s[TWO_N];
    __shared__ float sh_d[TWO_N];
    __shared__ float sh_p[256];
    int i   = blockIdx.x;
    int tid = threadIdx.x;
    float ti = T[i];
    for (int j = tid; j < TWO_N; j += 256) {
        sh_s[j] = S[(size_t)i * TWO_N + j];
        sh_d[j] = fabsf(ti - T[j]);
    }
    __syncthreads();

    float local = 0.0f;
    for (int k = tid; k < TWO_N; k += 256) {
        if (k == i) continue;
        float thr   = sh_d[k];
        float denom = 0.0f;
        #pragma unroll 4
        for (int j = 0; j < TWO_N; ++j) {
            float add = (sh_d[j] >= thr) ? sh_s[j] : 0.0f;
            denom += (j == i) ? 0.0f : add;
        }
        local += logf(sh_s[k]) - logf(denom);
    }
    sh_p[tid] = local;
    __syncthreads();
    for (int off = 128; off > 0; off >>= 1) {
        if (tid < off) sh_p[tid] += sh_p[tid + off];
        __syncthreads();
    }
    if (tid == 0) rowpart[i] = sh_p[0];
}

// --- Kernel 4: deterministic fixed-order final reduction
__global__ void k_final(const float* __restrict__ rowpart, float* __restrict__ out) {
    __shared__ float sh[256];
    int tid = threadIdx.x;
    sh[tid] = rowpart[tid] + rowpart[tid + 256];
    __syncthreads();
    for (int off = 128; off > 0; off >>= 1) {
        if (tid < off) sh[tid] += sh[tid + off];
        __syncthreads();
    }
    if (tid == 0) out[0] = -sh[0] / (float)(TWO_N * (TWO_N - 1));
}

extern "C" void kernel_launch(void* const* d_in, const int* in_sizes, int n_in,
                              void* d_out, int out_size, void* d_ws, size_t ws_size,
                              hipStream_t stream) {
    const float* E = (const float*)d_in[0];   // [512, 256] fp32
    const float* T = (const float*)d_in[1];   // [512, 1]   fp32
    float* out = (float*)d_out;

    char* ws = (char*)d_ws;
    __bf16* Ehi     = (__bf16*)(ws + OFF_EHI);
    __bf16* Elo     = (__bf16*)(ws + OFF_ELO);
    float*  nn      = (float*)(ws + OFF_NN);
    float*  S       = (float*)(ws + OFF_S);
    float*  rowpart = (float*)(ws + OFF_ROW);

    k_prep  <<<(TWO_N * DMODEL + 255) / 256, 256, 0, stream>>>(E, Ehi, Elo);
    k_norm  <<<2,   256, 0, stream>>>(E, nn);
    k_gemm_s<<<128, 256, 0, stream>>>(Ehi, Elo, nn, S);   // 1024 tiles / 8 waves per block
    k_denom <<<TWO_N, 256, 0, stream>>>(S, T, rowpart);
    k_final <<<1,   256, 0, stream>>>(rowpart, out);
}
